// TwoStageSelfgatingRNNARMA_57389353009772
// MI455X (gfx1250) — compile-verified, matrix operand order checked by reading, not run on hardware
//
#include <hip/hip_runtime.h>
#include <stdint.h>

// ---------------------------------------------------------------------------
// TwoStageSelfgatingRNNARMA for MI455X (gfx1250, wave32, WMMA)
//   Pass 0: f32 -> bf16 pre-conversion of x and the 4 weight matrices.
//   4 GEMMs: double-buffered LDS pipeline:
//              A tile: per-lane global_load_async_to_lds_b128 (ASYNCcnt)
//              B tile: Tensor Data Mover descriptor load     (TENSORcnt)
//            v_wmma_f32_16x16x32_bf16 compute, fused bias+activation.
//   Scan:    bandwidth-bound sequential recurrence along L per (b,d) channel,
//            writes Y directly as bf16 for the final GEMM.
// ---------------------------------------------------------------------------

#define TILE_M 128
#define TILE_N 128
#define TILE_K 32

typedef __attribute__((ext_vector_type(16))) __bf16 v16bf;
typedef __attribute__((ext_vector_type(8)))  __bf16 v8bf;
typedef __attribute__((ext_vector_type(8)))  float  v8f;
typedef unsigned int u32x4 __attribute__((ext_vector_type(4)));
typedef int          i32x4 __attribute__((ext_vector_type(4)));
typedef int          i32x8 __attribute__((ext_vector_type(8)));

#if __has_builtin(__builtin_amdgcn_tensor_load_to_lds)
#define HAVE_TDM 1
#else
#define HAVE_TDM 0
#endif

// Async per-stage outstanding ops per thread: A tile = 2 chunks; if no TDM,
// B tile adds 2 more.
#if HAVE_TDM
#define PER_STAGE_ASYNC 2
#else
#define PER_STAGE_ASYNC 4
#endif

// --- CDNA5 async global->LDS copy (16B per lane), tracked by ASYNCcnt -------
__device__ __forceinline__ void async_copy16(unsigned ldsOff, const void* g) {
  asm volatile("global_load_async_to_lds_b128 %0, %1, off"
               :: "v"(ldsOff), "v"(g)
               : "memory");
}

template <int N>
__device__ __forceinline__ void wait_async() {
#if __has_builtin(__builtin_amdgcn_s_wait_asynccnt)
  __builtin_amdgcn_s_wait_asynccnt((short)N);
#else
  asm volatile("s_wait_asynccnt %0" :: "i"(N) : "memory");
#endif
}

template <int N>
__device__ __forceinline__ void wait_tensor() {
#if __has_builtin(__builtin_amdgcn_s_wait_tensorcnt)
  __builtin_amdgcn_s_wait_tensorcnt((short)N);
#else
  asm volatile("s_wait_tensorcnt %0" :: "i"(N) : "memory");
#endif
}

// Generic-pointer low 32 bits == LDS byte offset on gfx1250 flat addressing.
__device__ __forceinline__ unsigned lds_off(const void* p) {
  return (unsigned)(uintptr_t)p;
}

// --- TDM: load a 2D tile (TILE_N rows x TILE_K cols, bf16, row stride K) ----
// D# per CDNA5 ISA 8.3/8.4: group0 = {flags, lds_addr, global_addr, type=2},
// group1 = {data_size=2B, tensor dims, tile dims, dim0 stride}.
__device__ __forceinline__ void tdm_load_tile(const void* gaddr, unsigned ldsOff,
                                              int rows, int Kdim) {
#if HAVE_TDM
  const unsigned long long ga = (unsigned long long)(uintptr_t)gaddr;
  u32x4 g0;
  g0[0] = 1u;                                   // count=1, user descriptor
  g0[1] = ldsOff;                               // lds_addr (bytes)
  g0[2] = (unsigned)(ga & 0xFFFFFFFFu);         // global_addr[31:0]
  g0[3] = (unsigned)((ga >> 32) & 0x01FFFFFFu)  // global_addr[56:32]
          | 0x80000000u;                        // type=2 ("image")
  i32x8 g1;
  g1[0] = 1 << 16;                              // data_size=1 (2 bytes)
  g1[1] = (Kdim & 0xFFFF) << 16;                // tensor_dim0[15:0]
  g1[2] = ((Kdim >> 16) & 0xFFFF) | ((rows & 0xFFFF) << 16);   // dim0 hi, dim1 lo
  g1[3] = ((rows >> 16) & 0xFFFF) | (TILE_K << 16);            // dim1 hi, tile_dim0
  g1[4] = TILE_N;                               // tile_dim1=128, tile_dim2=0
  g1[5] = Kdim;                                 // tensor_dim0_stride[31:0]
  g1[6] = 0;                                    // stride0 hi, stride1 lo
  g1[7] = 0;
  const i32x4 z4 = {0, 0, 0, 0};
#if defined(__clang_major__) && __clang_major__ >= 23
  const i32x8 z8 = {0, 0, 0, 0, 0, 0, 0, 0};
  __builtin_amdgcn_tensor_load_to_lds(g0, g1, z4, z4, z8, 0);
#else
  __builtin_amdgcn_tensor_load_to_lds(g0, g1, z4, z4, 0);
#endif
#endif
}

enum { EPI_EXP = 0, EPI_SIG = 1, EPI_MAK = 2, EPI_NONE = 3 };

template <int MODE>
__device__ __forceinline__ float epilogue(float v) {
  if (MODE == EPI_EXP) return __expf(v);
  if (MODE == EPI_SIG) return 1.0f / (1.0f + __expf(-v));
  // ma_k_activation: sigmoid(v / sqrt(D) * 0.02) = sigmoid(v * 6.25e-4)
  if (MODE == EPI_MAK) return 1.0f / (1.0f + __expf(-v * 6.25e-4f));
  return v;
}

// --- f32 -> bf16 pre-conversion (8 elems / thread, b128 in, b128 out) -------
__global__ __launch_bounds__(256) void f32_to_bf16_kernel(
    const float* __restrict__ in, __bf16* __restrict__ out, long n) {
  const long i = ((long)blockIdx.x * blockDim.x + threadIdx.x) * 8;
  if (i >= n) return;
  const float4 a = *(const float4*)(in + i);
  const float4 b = *(const float4*)(in + i + 4);
  v8bf r;
  r[0] = (__bf16)a.x; r[1] = (__bf16)a.y; r[2] = (__bf16)a.z; r[3] = (__bf16)a.w;
  r[4] = (__bf16)b.x; r[5] = (__bf16)b.y; r[6] = (__bf16)b.z; r[7] = (__bf16)b.w;
  *(v8bf*)(out + i) = r;
}

// C[m,n] = epilogue( sum_k A[m,k] * W[n,k] + bias[n] )   (torch Linear)
// A: (M,K) bf16 row-major, W: (N,K) bf16 row-major, out: (M,N) f32.
// 256 threads = 8 wave32 waves (4Mx2N); each wave: 32x64 out = 2x4 WMMA tiles.
// Double-buffered LDS: tile k+1 streams in (async + TDM) under tile k's WMMAs.
template <int MODE>
__global__ __launch_bounds__(256) void gemm_bf16_wmma(
    const __bf16* __restrict__ A, const __bf16* __restrict__ W,
    const float* __restrict__ bias, float* __restrict__ out,
    int M, int N, int K) {
  __shared__ __bf16 Alds[2][TILE_M * TILE_K];  // 2 x 8 KB
  __shared__ __bf16 Blds[2][TILE_N * TILE_K];  // 2 x 8 KB

  const int tid   = threadIdx.x;
  const int lane  = tid & 31;
  const int wv    = tid >> 5;   // 0..7
  const int waveM = wv >> 1;    // 0..3 -> 32 rows each
  const int waveN = wv & 1;     // 0..1 -> 64 cols each
  const int mBase = blockIdx.y * TILE_M;
  const int nBase = blockIdx.x * TILE_N;

  const v8f vzero = {};
  v8f acc[2][4];
#pragma unroll
  for (int i = 0; i < 2; ++i)
#pragma unroll
    for (int j = 0; j < 4; ++j) acc[i][j] = vzero;

  // Stage one K-tile (A via per-lane async chunks, B via one TDM descriptor).
  auto stage = [&](int k0, int buf) {
#pragma unroll
    for (int i = 0; i < 2; ++i) {
      const int c   = tid + i * 256;  // 0..511
      const int row = c >> 2;         // 0..127
      const int sub = c & 3;          // 16B chunk within 64B row
      async_copy16(lds_off(&Alds[buf][row * TILE_K + sub * 8]),
                   A + (size_t)(mBase + row) * K + k0 + sub * 8);
#if !HAVE_TDM
      async_copy16(lds_off(&Blds[buf][row * TILE_K + sub * 8]),
                   W + (size_t)(nBase + row) * K + k0 + sub * 8);
#endif
    }
#if HAVE_TDM
    if (wv == 0) {  // one TDM tile-load per block per K-step
      tdm_load_tile(W + (size_t)nBase * K + k0, lds_off(&Blds[buf][0]), N, K);
    }
#endif
  };

  const int nk = K / TILE_K;
  stage(0, 0);

  for (int kt = 0; kt < nk; ++kt) {
    const int buf = kt & 1;
    if (kt + 1 < nk) {
      stage((kt + 1) * TILE_K, buf ^ 1);  // prefetch next tile into other buf
      wait_async<PER_STAGE_ASYNC>();      // tile kt's async copies done
      if (HAVE_TDM && wv == 0) wait_tensor<1>();
    } else {
      wait_async<0>();
      if (HAVE_TDM && wv == 0) wait_tensor<0>();
    }
    __syncthreads();  // publish tile kt to all waves

    // Fragment gather (contiguous k order used here for A and B: lanes 0-15 <-> K 0..15, lanes 16-31 <->
    // K 16..31; M/N = lane & 15). One 32B contiguous LDS read per fragment.
    const int kSel = (lane >> 4) * 16;
    const int mn   = lane & 15;
    v16bf afrag[2], bfrag[4];
#pragma unroll
    for (int i = 0; i < 2; ++i) {
      const int m = waveM * 32 + i * 16 + mn;
      afrag[i] = *(const v16bf*)(&Alds[buf][m * TILE_K + kSel]);
    }
#pragma unroll
    for (int j = 0; j < 4; ++j) {
      const int n = waveN * 64 + j * 16 + mn;
      bfrag[j] = *(const v16bf*)(&Blds[buf][n * TILE_K + kSel]);
    }

#pragma unroll
    for (int i = 0; i < 2; ++i)
#pragma unroll
      for (int j = 0; j < 4; ++j)
        acc[i][j] = __builtin_amdgcn_wmma_f32_16x16x32_bf16(
            false, afrag[i], false, bfrag[j], (short)0, acc[i][j], false,
            false);

    __syncthreads();  // all reads of buf done before it is restaged
  }

  // Epilogue. C/D layout: VGPR r -> M = r + 8*(lane>=16), N = lane & 15.
  const int colLane = lane & 15;
  const int mHalf   = (lane >> 4) * 8;
#pragma unroll
  for (int j = 0; j < 4; ++j) {
    const int col = nBase + waveN * 64 + j * 16 + colLane;
    const float bv = bias[col];
#pragma unroll
    for (int i = 0; i < 2; ++i) {
      const int rowBase = mBase + waveM * 32 + i * 16 + mHalf;
#pragma unroll
      for (int r = 0; r < 8; ++r) {
        out[(size_t)(rowBase + r) * N + col] = epilogue<MODE>(acc[i][j][r] + bv);
      }
    }
  }
}

// Sequential scan along L per (b,d) channel; coalesced over d.
//   kc   = cumsum(K)+1e-6 ; S = cumsum(K*V)/kc ; O_ar = S*Q
//   E[t] = V[t] - O_ar[t-1] (0 at t=0)
//   S2  += ma_k_act(lin(x[t-1])) * E[t]  (stored K2 is unshifted; use prev)
//   O_ma = S2 * ma_q_act(Q[t-1]) ; Y = O_ar + O_ma (written as bf16)
__global__ __launch_bounds__(256) void scan_kernel(
    const float* __restrict__ x, const float* __restrict__ Kk,
    const float* __restrict__ Qq, const float* __restrict__ K2a,
    __bf16* __restrict__ Yb, int Bb, int L, int Dd) {
  const int idx = blockIdx.x * blockDim.x + threadIdx.x;  // b*D + d
  if (idx >= Bb * Dd) return;
  const int b = idx / Dd;
  const int d = idx - b * Dd;
  const size_t base = (size_t)b * L * Dd + d;

  float kc = 1e-6f, snum = 0.0f, s2 = 0.0f;
  float oar_prev = 0.0f, k2_prev = 0.0f, q_prev = 0.0f;

  for (int t = 0; t < L; ++t) {
    const size_t off = base + (size_t)t * Dd;
    if (t + 16 < L) {  // global_prefetch_b8 hints for the streaming reads
      __builtin_prefetch(Kk + off + (size_t)16 * Dd, 0, 0);
      __builtin_prefetch(x + off + (size_t)16 * Dd, 0, 0);
      __builtin_prefetch(Qq + off + (size_t)16 * Dd, 0, 0);
      __builtin_prefetch(K2a + off + (size_t)16 * Dd, 0, 0);
    }
    const float k  = Kk[off];
    const float q  = Qq[off];
    const float k2 = K2a[off];
    const float v  = x[off];

    kc += k;
    snum += k * v;
    const float oar = (snum / kc) * q;

    const float E = (t == 0) ? 0.0f : (v - oar_prev);
    s2 += k2_prev * E;
    // ma_q_activation of previous Q: -leaky_relu(-q/sqrt(D), 0.02)
    const float qa = -q_prev * (1.0f / 32.0f);
    const float q2 = -((qa >= 0.0f) ? qa : 0.02f * qa);
    const float oma = (t == 0) ? 0.0f : s2 * q2;

    Yb[off] = (__bf16)(oar + oma);

    oar_prev = oar;
    k2_prev  = k2;
    q_prev   = q;
  }
}

extern "C" void kernel_launch(void* const* d_in, const int* in_sizes, int n_in,
                              void* d_out, int out_size, void* d_ws,
                              size_t ws_size, hipStream_t stream) {
  const float* x     = (const float*)d_in[0];
  const float* Wq1   = (const float*)d_in[1];
  const float* bq1   = (const float*)d_in[2];
  const float* Wk1   = (const float*)d_in[3];
  const float* bk1   = (const float*)d_in[4];
  const float* Wk2   = (const float*)d_in[5];
  const float* bk2   = (const float*)d_in[6];
  const float* Wproj = (const float*)d_in[7];
  const float* bproj = (const float*)d_in[8];

  const int Dd = 1024;
  const int L  = 4096;
  const int M  = in_sizes[0] / Dd;  // B*L = 16384
  const int Bb = M / L;             // 4

  const size_t plane = (size_t)M * Dd;   // activation plane elements
  const size_t wsz   = (size_t)Dd * Dd;  // weight elements

  // Workspace layout (f32 planes first, then bf16 buffers).
  char* ws = (char*)d_ws;
  float*  wsK   = (float*)ws;                 ws += plane * sizeof(float);
  float*  wsQ   = (float*)ws;                 ws += plane * sizeof(float);
  float*  wsK2  = (float*)ws;                 ws += plane * sizeof(float);
  __bf16* xb    = (__bf16*)ws;                ws += plane * sizeof(__bf16);
  __bf16* Yb    = (__bf16*)ws;                ws += plane * sizeof(__bf16);
  __bf16* Wq1b  = (__bf16*)ws;                ws += wsz * sizeof(__bf16);
  __bf16* Wk1b  = (__bf16*)ws;                ws += wsz * sizeof(__bf16);
  __bf16* Wk2b  = (__bf16*)ws;                ws += wsz * sizeof(__bf16);
  __bf16* Wprb  = (__bf16*)ws;

  dim3 blk(256);

  // Pass 0: pre-convert operands to bf16 (once).
  {
    const long nx = (long)plane;
    f32_to_bf16_kernel<<<dim3((unsigned)((nx / 8 + 255) / 256)), blk, 0,
                         stream>>>(x, xb, nx);
    const long nw = (long)wsz;
    const dim3 gw((unsigned)((nw / 8 + 255) / 256));
    f32_to_bf16_kernel<<<gw, blk, 0, stream>>>(Wq1, Wq1b, nw);
    f32_to_bf16_kernel<<<gw, blk, 0, stream>>>(Wk1, Wk1b, nw);
    f32_to_bf16_kernel<<<gw, blk, 0, stream>>>(Wk2, Wk2b, nw);
    f32_to_bf16_kernel<<<gw, blk, 0, stream>>>(Wproj, Wprb, nw);
  }

  const dim3 grd(Dd / TILE_N, M / TILE_M);  // 8 x 128 = 1024 blocks

  gemm_bf16_wmma<EPI_EXP><<<grd, blk, 0, stream>>>(xb, Wk1b, bk1, wsK, M, Dd, Dd);
  gemm_bf16_wmma<EPI_SIG><<<grd, blk, 0, stream>>>(xb, Wq1b, bq1, wsQ, M, Dd, Dd);
  gemm_bf16_wmma<EPI_MAK><<<grd, blk, 0, stream>>>(xb, Wk2b, bk2, wsK2, M, Dd, Dd);

  const int nchan = Bb * Dd;  // 4096 channels
  scan_kernel<<<dim3((nchan + 255) / 256), blk, 0, stream>>>(x, wsK, wsQ, wsK2,
                                                             Yb, Bb, L, Dd);

  gemm_bf16_wmma<EPI_NONE><<<grd, blk, 0, stream>>>(Yb, Wprb, bproj,
                                                    (float*)d_out, M, Dd, Dd);
}
